// Qwen3MoeSparseMoeBlock_16690242912515
// MI455X (gfx1250) — matrix operand order repaired and checked by reading, not compiled, not run on hardware
//
#include <hip/hip_runtime.h>

// ---------------------------------------------------------------------------
// Qwen3-style MoE block, specialized for T=2048,H=2048,E=8,K=8,I=768,CAP=512.
// K==E: every token routes to every expert with weight softmax(logits);
// capacity dropping keeps only tokens 0..511 (slot-in-expert == token id).
// out[t<512] = sum_e p[t][e] * down_e(silu(gate_e(x_t)) * up_e(x_t)); else 0.
//
// bf16 WMMA (16x16x32) with f32 accum; f32 weights are DMA'd raw into LDS
// with GLOBAL_LOAD_ASYNC_TO_LDS_B32 (per-lane LDS addressing performs the
// [K][N]->[N][K] transpose), double-buffered, converted to bf16 at
// fragment-build time. All B fragments are materialized before the WMMA
// burst to avoid the WMMA->VALU WAR hazard NOPs (ISA 7.12.1).
// ---------------------------------------------------------------------------

#define T_TOK 2048
#define H_DIM 2048
#define E_NUM 8
#define I_DIM 768
#define CAP   512

typedef unsigned short u16;
typedef u16   v8u16  __attribute__((ext_vector_type(8)));
typedef u16   v16u16 __attribute__((ext_vector_type(16)));
typedef __bf16 v16bf __attribute__((ext_vector_type(16)));
typedef float  v8f   __attribute__((ext_vector_type(8)));
typedef float  v4f   __attribute__((ext_vector_type(4)));

#if defined(__has_builtin)
#if __has_builtin(__builtin_amdgcn_global_load_async_to_lds_b32) && \
    __has_builtin(__builtin_amdgcn_s_wait_asynccnt)
#define USE_ASYNC_LDS 1
#endif
#endif
#ifndef USE_ASYNC_LDS
#define USE_ASYNC_LDS 0
#endif

typedef __attribute__((address_space(1))) int* gbl_i32p;
typedef __attribute__((address_space(3))) int* lds_i32p;

__device__ __forceinline__ u16 f2bf(float f) {
  return __builtin_bit_cast(u16, (__bf16)f);   // v_cvt_pk_bf16_f32 (RNE)
}

// stage one f32 element global -> LDS (async DMA when available)
__device__ __forceinline__ void stage_b32(const float* g, float* l) {
#if USE_ASYNC_LDS
  __builtin_amdgcn_global_load_async_to_lds_b32(
      (gbl_i32p)(g), (lds_i32p)(l), 0, 0);
#else
  *l = *g;   // ds_store path; __syncthreads() provides the dscnt wait
#endif
}

__device__ __forceinline__ void wait_stage() {
#if USE_ASYNC_LDS
  __builtin_amdgcn_s_wait_asynccnt(0);
#endif
}

// stage a 32(K) x 64(N) f32 tile, transposed into LDS as [col][32]
__device__ __forceinline__ void stage_tile(
    float* __restrict__ dst, const float* __restrict__ src, int ld, int tid) {
#pragma unroll
  for (int j = 0; j < 16; ++j) {
    int idx = j * 128 + tid;          // global-coalesced: lane-consecutive cols
    int row = idx >> 6, col = idx & 63;
    stage_b32(src + (size_t)row * ld + col, dst + col * 32 + row);
  }
}

// read 16 contiguous f32 from LDS, convert to a bf16 WMMA operand half-pair
__device__ __forceinline__ v16u16 cvt_frag(const float* __restrict__ lf) {
  const v4f* p = (const v4f*)lf;      // 4 x ds_load_b128
  v4f x0 = p[0], x1 = p[1], x2 = p[2], x3 = p[3];
  v16u16 r;
#pragma unroll
  for (int i = 0; i < 4; ++i) {
    r[i]      = f2bf(x0[i]);
    r[4 + i]  = f2bf(x1[i]);
    r[8 + i]  = f2bf(x2[i]);
    r[12 + i] = f2bf(x3[i]);
  }
  return r;
}

__device__ __forceinline__ v8f bf16_wmma(v16u16 a, v16u16 b, v8f c) {
  return __builtin_amdgcn_wmma_f32_16x16x32_bf16(
      /*neg_a=*/false, __builtin_bit_cast(v16bf, a),
      /*neg_b=*/false, __builtin_bit_cast(v16bf, b),
      /*c_mod=*/(short)0, c, /*reuse_a=*/false, /*reuse_b=*/false);
}

__device__ __forceinline__ v16u16 mk_afrag(v8u16 a0, v8u16 a1) {
  v16u16 a;
#pragma unroll
  for (int i = 0; i < 8; ++i) { a[i] = a0[i]; a[8 + i] = a1[i]; }
  return a;
}

// ---------------- router: logits -> softmax (== final combine weights) -----
__global__ __launch_bounds__(32) void router_kernel(
    const float* __restrict__ x, const float* __restrict__ wr,
    float* __restrict__ probs) {
  int t = blockIdx.x;          // token 0..511
  int lane = threadIdx.x;      // wave32
  float acc[E_NUM];
#pragma unroll
  for (int e = 0; e < E_NUM; ++e) acc[e] = 0.f;
  for (int h = lane; h < H_DIM; h += 32) {
    float xv = x[(size_t)t * H_DIM + h];
#pragma unroll
    for (int e = 0; e < E_NUM; ++e) acc[e] += xv * wr[(size_t)e * H_DIM + h];
  }
#pragma unroll
  for (int e = 0; e < E_NUM; ++e)
#pragma unroll
    for (int off = 16; off > 0; off >>= 1)
      acc[e] += __shfl_xor(acc[e], off, 32);
  if (lane == 0) {
    float m = acc[0];
#pragma unroll
    for (int e = 1; e < E_NUM; ++e) m = fmaxf(m, acc[e]);
    float p[E_NUM], s = 0.f;
#pragma unroll
    for (int e = 0; e < E_NUM; ++e) { p[e] = __expf(acc[e] - m); s += p[e]; }
    float inv = __builtin_amdgcn_rcpf(s);  // top-K==E renorm is identity
#pragma unroll
    for (int e = 0; e < E_NUM; ++e) probs[t * E_NUM + e] = p[e] * inv;
  }
}

// ---------------- x[0:512] f32 -> bf16 -------------------------------------
__global__ __launch_bounds__(256) void cvt_x_kernel(
    const float* __restrict__ x, u16* __restrict__ xb) {
  int i = blockIdx.x * 256 + threadIdx.x;   // grid covers exactly CAP*H
  xb[i] = f2bf(x[i]);
}

// ---------------- GEMM1: Xb[512x2048] x {Wg,Wu}_e[2048x768], SwiGLU --------
// 128 threads = 4 waves; block tile 64(M) x 64(N); wave owns 16 rows.
// A fragments straight from global (contiguous k = 16*hi + s, 2 x b128/lane).
// B tiles: raw f32, async-DMA'd transposed into double-buffered LDS;
// one barrier + one asynccnt wait per K-step.
__global__ __launch_bounds__(128) void gemm1_kernel(
    const u16* __restrict__ xb, const float* __restrict__ wg,
    const float* __restrict__ wu, u16* __restrict__ hact) {
  __shared__ float lgf[2][32 * 64];
  __shared__ float luf[2][32 * 64];
  const int e     = blockIdx.z;
  const int m_blk = blockIdx.x * 64;
  const int n_blk = blockIdx.y * 64;
  const int tid  = threadIdx.x;
  const int wave = tid >> 5, lane = tid & 31;
  const int hi = lane >> 4, lm = lane & 15;
  const int m_base = m_blk + wave * 16;
  const float* wgE = wg + (size_t)e * H_DIM * I_DIM + n_blk;
  const float* wuE = wu + (size_t)e * H_DIM * I_DIM + n_blk;

  const v8f zero = {};
  v8f accg[4], accu[4];
#pragma unroll
  for (int i = 0; i < 4; ++i) { accg[i] = zero; accu[i] = zero; }

  // prologue: stage tile 0, preload A fragment 0
  stage_tile(lgf[0], wgE, I_DIM, tid);
  stage_tile(luf[0], wuE, I_DIM, tid);
  const u16* arow = xb + (size_t)(m_base + lm) * H_DIM + hi * 16;
  v8u16 a0 = *(const v8u16*)(arow);
  v8u16 a1 = *(const v8u16*)(arow + 8);

  const int KT = H_DIM / 32;   // 64
  for (int kt = 0; kt < KT; ++kt) {
    const int cur = kt & 1, nxt = cur ^ 1;
    wait_stage();        // my buf[cur] DMAs landed
    __syncthreads();     // all waves: cur published; prior reads of nxt done
    if (kt + 1 < KT) {   // DMA tile kt+1 while computing tile kt
      stage_tile(lgf[nxt], wgE + (size_t)(kt + 1) * 32 * I_DIM, I_DIM, tid);
      stage_tile(luf[nxt], wuE + (size_t)(kt + 1) * 32 * I_DIM, I_DIM, tid);
    }
    v16u16 afrag = mk_afrag(a0, a1);
    if (kt + 1 < KT) {   // pipeline next A fragment
      const u16* an = arow + (kt + 1) * 32;
      a0 = *(const v8u16*)(an);
      a1 = *(const v8u16*)(an + 8);
    }
    // materialize all B fragments first (disjoint regs), then WMMA burst:
    // avoids WMMA->VALU WAR hazard nops between consecutive WMMAs.
    v16u16 bg[4], bu[4];
#pragma unroll
    for (int ns = 0; ns < 4; ++ns) {
      const int c = (ns * 16 + lm) * 32 + hi * 16;
      bg[ns] = cvt_frag(&lgf[cur][c]);
      bu[ns] = cvt_frag(&luf[cur][c]);
    }
#pragma unroll
    for (int ns = 0; ns < 4; ++ns) {
      accg[ns] = bf16_wmma(afrag, bg[ns], accg[ns]);
      accu[ns] = bf16_wmma(afrag, bu[ns], accu[ns]);
    }
  }

  // epilogue: h = silu(g) * u  -> bf16 hact[e][row][col]
  u16* hE = hact + (size_t)e * CAP * I_DIM;
#pragma unroll
  for (int ns = 0; ns < 4; ++ns) {
    const int col = n_blk + ns * 16 + lm;
#pragma unroll
    for (int r = 0; r < 8; ++r) {
      const int row = m_base + r + 8 * hi;   // C/D layout: vgpr r, lane-half
      float g = accg[ns][r], u = accu[ns][r];
      float h = g * __builtin_amdgcn_rcpf(1.f + __expf(-g)) * u;
      hE[(size_t)row * I_DIM + col] = f2bf(h);
    }
  }
}

// ---------------- GEMM2: sum_e p[:,e] * (hact_e[512x768] x Wd_e[768x2048]) -
__global__ __launch_bounds__(128) void gemm2_kernel(
    const u16* __restrict__ hact, const float* __restrict__ wd,
    const float* __restrict__ probs, float* __restrict__ out) {
  __shared__ float lbf[2][32 * 64];
  const int m_blk = blockIdx.x * 64;
  const int n_blk = blockIdx.y * 64;
  const int tid  = threadIdx.x;
  const int wave = tid >> 5, lane = tid & 31;
  const int hi = lane >> 4, lm = lane & 15;
  const int m_base = m_blk + wave * 16;

  const v8f zero = {};
  v8f acco[4];
#pragma unroll
  for (int i = 0; i < 4; ++i) acco[i] = zero;

  for (int e = 0; e < E_NUM; ++e) {
    const u16*   hE  = hact + (size_t)e * CAP * I_DIM;
    const float* wdE = wd + (size_t)e * I_DIM * H_DIM + n_blk;
    v8f acc[4];
#pragma unroll
    for (int i = 0; i < 4; ++i) acc[i] = zero;

    stage_tile(lbf[0], wdE, H_DIM, tid);
    const u16* arow = hE + (size_t)(m_base + lm) * I_DIM + hi * 16;
    v8u16 a0 = *(const v8u16*)(arow);
    v8u16 a1 = *(const v8u16*)(arow + 8);

    const int KT = I_DIM / 32;   // 24
    for (int kt = 0; kt < KT; ++kt) {
      const int cur = kt & 1, nxt = cur ^ 1;
      wait_stage();
      __syncthreads();
      if (kt + 1 < KT)
        stage_tile(lbf[nxt], wdE + (size_t)(kt + 1) * 32 * H_DIM, H_DIM, tid);
      v16u16 afrag = mk_afrag(a0, a1);
      if (kt + 1 < KT) {
        const u16* an = arow + (kt + 1) * 32;
        a0 = *(const v8u16*)(an);
        a1 = *(const v8u16*)(an + 8);
      }
      v16u16 bfrag[4];
#pragma unroll
      for (int ns = 0; ns < 4; ++ns)
        bfrag[ns] = cvt_frag(&lbf[cur][(ns * 16 + lm) * 32 + hi * 16]);
#pragma unroll
      for (int ns = 0; ns < 4; ++ns)
        acc[ns] = bf16_wmma(afrag, bfrag[ns], acc[ns]);
    }
    __syncthreads();   // buffers drained before next expert's prologue stage

    // combine with router weight (per output row)
    float pr[8];
#pragma unroll
    for (int r = 0; r < 8; ++r)
      pr[r] = probs[(size_t)(m_base + r + 8 * hi) * E_NUM + e];
#pragma unroll
    for (int ns = 0; ns < 4; ++ns)
#pragma unroll
      for (int r = 0; r < 8; ++r) acco[ns][r] += pr[r] * acc[ns][r];
  }

#pragma unroll
  for (int ns = 0; ns < 4; ++ns) {
    const int col = n_blk + ns * 16 + lm;
#pragma unroll
    for (int r = 0; r < 8; ++r) {
      const int row = m_base + r + 8 * hi;
      out[(size_t)row * H_DIM + col] = acco[ns][r];
    }
  }
}

// ---------------------------------------------------------------------------
extern "C" void kernel_launch(void* const* d_in, const int* in_sizes, int n_in,
                              void* d_out, int out_size, void* d_ws,
                              size_t ws_size, hipStream_t stream) {
  (void)in_sizes; (void)n_in; (void)out_size; (void)ws_size;
  const float* x  = (const float*)d_in[0];   // [T,H]
  const float* wr = (const float*)d_in[1];   // [E,H]
  const float* wg = (const float*)d_in[2];   // [E,H,I]
  const float* wu = (const float*)d_in[3];   // [E,H,I]
  const float* wd = (const float*)d_in[4];   // [E,I,H]
  float* out = (float*)d_out;                // [T,H] f32

  char* ws = (char*)d_ws;
  u16*   xb    = (u16*)ws;                                         // 2 MB
  float* probs = (float*)(ws + (size_t)CAP * H_DIM * sizeof(u16)); // 16 KB
  u16*   hact  = (u16*)(ws + (size_t)CAP * H_DIM * sizeof(u16)
                           + (size_t)CAP * E_NUM * sizeof(float)); // 6 MB

  // tokens >= CAP are dropped by capacity -> exact zeros
  (void)hipMemsetAsync(out, 0, (size_t)T_TOK * H_DIM * sizeof(float), stream);

  router_kernel<<<CAP, 32, 0, stream>>>(x, wr, probs);
  cvt_x_kernel<<<(CAP * H_DIM) / 256, 256, 0, stream>>>(x, xb);
  gemm1_kernel<<<dim3(CAP / 64, I_DIM / 64, E_NUM), 128, 0, stream>>>(
      xb, wg, wu, hact);
  gemm2_kernel<<<dim3(CAP / 64, H_DIM / 64), 128, 0, stream>>>(
      hact, wd, probs, out);
}
